// CustomKernelDense_70506183131345
// MI455X (gfx1250) — matrix operand order repaired and checked by reading, not compiled, not run on hardware
//
#include <hip/hip_runtime.h>
#include <hip/hip_bf16.h>

// ---------------------------------------------------------------------------
// RBF dense layer: out[b,u] = exp(-(|x_b|^2 + |k_u|^2 - 2 * (x @ K)[b,u])) + bias[u]
// B=8192, D=512, U=4096, fp32 in/out.
//
// Pipeline:
//   1) row/col squared norms -> ws
//   2) one-time fp32 -> bf16 hi/lo decomposition of X (row-major) and K^T
//      (n-major) -> ws  (removes all conversion VALU from the GEMM hot loop)
//   3) GEMM with bf16 WMMA, 3-product split precision (hi*hi + hi*lo + lo*hi),
//      double-buffered LDS tiles filled with GLOBAL_LOAD_ASYNC_TO_LDS_B128
//      (ASYNCcnt) when available, fused exp/bias epilogue.
// ---------------------------------------------------------------------------

typedef __attribute__((ext_vector_type(16))) __bf16 v16bf;
typedef __attribute__((ext_vector_type(8)))  float  v8f;
typedef __attribute__((ext_vector_type(4)))  int    v4i;

union FragBF {
    v16bf v;
    uint4 q[2];
};

__device__ __forceinline__ unsigned short f2bf_rne(float f) {
    unsigned u = __float_as_uint(f);
    unsigned r = u + 0x7FFFu + ((u >> 16) & 1u);   // round-to-nearest-even
    return (unsigned short)(r >> 16);
}
__device__ __forceinline__ float bf2f(unsigned short h) {
    return __uint_as_float(((unsigned)h) << 16);
}

#define B_DIM 8192
#define D_DIM 512
#define U_DIM 4096

#define BM 128
#define BN 128
#define BK 32
#define LDAS 40   // padded LDS stride (bf16 elems): 80 B, keeps 16-B alignment

// -------------------- async global->LDS copy (CDNA5) -----------------------
#if defined(__has_builtin)
#  if __has_builtin(__builtin_amdgcn_global_load_async_to_lds_b128)
#    define HAVE_ASYNC_LDS 1
#  endif
#endif
#ifndef HAVE_ASYNC_LDS
#  define HAVE_ASYNC_LDS 0
#endif

// Builtin signature (from hipcc diagnostic): first arg is AS1 int4*, LDS arg
// is the AS3 counterpart. Cast via integer round-trip: a generic LDS pointer's
// low 32 bits ARE the LDS offset on AMDGPU, so this is semantically correct.
typedef __attribute__((address_space(1))) v4i g_v4i;
typedef __attribute__((address_space(3))) v4i l_v4i;

__device__ __forceinline__ void copy16_g2lds(const unsigned short* g, unsigned short* l) {
#if HAVE_ASYNC_LDS
    __builtin_amdgcn_global_load_async_to_lds_b128(
        (g_v4i*)(unsigned long long)(uintptr_t)g,
        (l_v4i*)(unsigned)(uintptr_t)l,
        0, 0);
#else
    uint4 v = *reinterpret_cast<const uint4*>(g);
    *reinterpret_cast<uint4*>(l) = v;
#endif
}

__device__ __forceinline__ void wait_async_zero() {
#if HAVE_ASYNC_LDS
#  if defined(__has_builtin) && __has_builtin(__builtin_amdgcn_s_wait_asynccnt)
    __builtin_amdgcn_s_wait_asynccnt(0);
#  else
    asm volatile("s_wait_asynccnt 0" ::: "memory");
#  endif
#endif
}

// ---------------------------------------------------------------------------
// |x_b|^2 : one wave32 per row (coalesced row reads + shuffle reduction)
// ---------------------------------------------------------------------------
__global__ void row_sq_norms(const float* __restrict__ x, float* __restrict__ xsq) {
    int wave = (blockIdx.x * blockDim.x + threadIdx.x) >> 5;
    int lane = threadIdx.x & 31;
    if (wave >= B_DIM) return;
    const float* row = x + (size_t)wave * D_DIM;
    float s = 0.f;
    for (int j = lane; j < D_DIM; j += 32) {
        float v = row[j];
        s = fmaf(v, v, s);
    }
    for (int off = 16; off > 0; off >>= 1) s += __shfl_down(s, off, 32);
    if (lane == 0) xsq[wave] = s;
}

// ---------------------------------------------------------------------------
// |k_u|^2 : one thread per column (consecutive threads -> coalesced)
// ---------------------------------------------------------------------------
__global__ void col_sq_norms(const float* __restrict__ k, float* __restrict__ ksq) {
    int u = blockIdx.x * blockDim.x + threadIdx.x;
    if (u >= U_DIM) return;
    float s = 0.f;
    for (int d = 0; d < D_DIM; ++d) {
        float v = k[(size_t)d * U_DIM + u];
        s = fmaf(v, v, s);
    }
    ksq[u] = s;
}

// ---------------------------------------------------------------------------
// One-time fp32 -> bf16 hi/lo split of X, same row-major (8192 x 512) layout.
// ---------------------------------------------------------------------------
__global__ void convert_x(const float* __restrict__ X,
                          unsigned short* __restrict__ Xhi,
                          unsigned short* __restrict__ Xlo) {
    size_t i = (size_t)blockIdx.x * blockDim.x + threadIdx.x;  // float4 index
    const float4 g = reinterpret_cast<const float4*>(X)[i];
    float fv[4] = {g.x, g.y, g.z, g.w};
    ushort4 h, l;
    unsigned short* hp = &h.x;
    unsigned short* lp = &l.x;
    #pragma unroll
    for (int e = 0; e < 4; ++e) {
        unsigned short hi = f2bf_rne(fv[e]);
        hp[e] = hi;
        lp[e] = f2bf_rne(fv[e] - bf2f(hi));
    }
    reinterpret_cast<ushort4*>(Xhi)[i] = h;
    reinterpret_cast<ushort4*>(Xlo)[i] = l;
}

// ---------------------------------------------------------------------------
// One-time fp32 -> bf16 hi/lo split of K, stored TRANSPOSED: KT[u][d]
// (n-major, k contiguous) so the GEMM B-tile staging is contiguous 16-B copies.
// ---------------------------------------------------------------------------
__global__ void convert_k_t(const float* __restrict__ Kmat,
                            unsigned short* __restrict__ KThi,
                            unsigned short* __restrict__ KTlo) {
    size_t i = (size_t)blockIdx.x * blockDim.x + threadIdx.x;  // float4 index
    int d  = (int)(i >> 10);          // / (4096/4)
    int uq = (int)(i & 1023);
    const float4 g = reinterpret_cast<const float4*>(Kmat + (size_t)d * U_DIM)[uq];
    float fv[4] = {g.x, g.y, g.z, g.w};
    #pragma unroll
    for (int e = 0; e < 4; ++e) {
        int u = uq * 4 + e;
        unsigned short hi = f2bf_rne(fv[e]);
        KThi[(size_t)u * D_DIM + d] = hi;
        KTlo[(size_t)u * D_DIM + d] = f2bf_rne(fv[e] - bf2f(hi));
    }
}

// ---------------------------------------------------------------------------
// Main fused GEMM + RBF epilogue.
// Block: 256 threads (8 wave32), tile 128x128, BK=32 (one bf16 WMMA K-step).
// Wave grid 2(M) x 4(N); each wave owns 4x2 16x16 accumulator tiles.
// Double-buffered LDS, async global->LDS staging.
// ---------------------------------------------------------------------------
__global__ __launch_bounds__(256, 1)
void rbf_gemm(const unsigned short* __restrict__ Xhi,
              const unsigned short* __restrict__ Xlo,
              const unsigned short* __restrict__ KThi,
              const unsigned short* __restrict__ KTlo,
              const float* __restrict__ bias, const float* __restrict__ xsq,
              const float* __restrict__ ksq, float* __restrict__ out) {
    __shared__ __align__(16) unsigned short AhiS[2][BM * LDAS];
    __shared__ __align__(16) unsigned short AloS[2][BM * LDAS];
    __shared__ __align__(16) unsigned short BhiS[2][BN * LDAS];   // transposed: [n][k]
    __shared__ __align__(16) unsigned short BloS[2][BN * LDAS];

    const int tid   = threadIdx.x;
    const int lane  = tid & 31;
    const int wave  = tid >> 5;
    const int waveM = wave >> 2;   // 0..1
    const int waveN = wave & 3;    // 0..3
    const int m0 = blockIdx.y * BM;
    const int n0 = blockIdx.x * BN;

    v8f acc[4][2];
    const v8f vzero = {0.f, 0.f, 0.f, 0.f, 0.f, 0.f, 0.f, 0.f};
    for (int i = 0; i < 4; ++i)
        for (int j = 0; j < 2; ++j)
            acc[i][j] = vzero;

    // Stage one 128x32 (A) + 128x32 (B^T) bf16 hi/lo tile set into LDS buffer
    // `buf` for K-slab starting at k0. 16-B chunks; 8 copies per thread.
    auto stage = [&](int buf, int k0) {
        #pragma unroll
        for (int j = 0; j < 2; ++j) {
            int c   = tid + 256 * j;   // 0..511 : 16-B chunk index
            int row = c >> 2;          // 0..127 (A row, or B column n)
            int kq  = c & 3;           // 0..3   (which 8-elem chunk of the 32-wide k slab)
            int ldsOff = row * LDAS + kq * 8;
            size_t gOffA = (size_t)(m0 + row) * D_DIM + k0 + kq * 8;
            size_t gOffB = (size_t)(n0 + row) * D_DIM + k0 + kq * 8;
            copy16_g2lds(Xhi  + gOffA, &AhiS[buf][ldsOff]);
            copy16_g2lds(Xlo  + gOffA, &AloS[buf][ldsOff]);
            copy16_g2lds(KThi + gOffB, &BhiS[buf][ldsOff]);
            copy16_g2lds(KTlo + gOffB, &BloS[buf][ldsOff]);
        }
    };

    stage(0, 0);
    int cur = 0;

    for (int k0 = 0; k0 < D_DIM; k0 += BK) {
        wait_async_zero();   // our async copies for buf `cur` have landed
        __syncthreads();     // everyone's copies landed; prev-iter LDS reads done

        if (k0 + BK < D_DIM) stage(cur ^ 1, k0 + BK);   // overlap next slab

        // ---- build WMMA fragments straight from LDS (16-B ds_load_b128 pairs) ----
        // A 16x32 bf16 layout: lanes 0-15 -> M=lane, K 0..7 (v0-3) & 16..23 (v4-7);
        //                      lanes 16-31 -> M=lane-16, K 8..15 & 24..31.
        FragBF ahi[4], alo[4], bhi[2], blo[2];
        const int mrow = lane & 15;
        const int kbA  = (lane < 16) ? 0 : 8;
        #pragma unroll
        for (int tm = 0; tm < 4; ++tm) {
            const unsigned short* pa = &AhiS[cur][(waveM * 64 + tm * 16 + mrow) * LDAS + kbA];
            ahi[tm].q[0] = *reinterpret_cast<const uint4*>(pa);
            ahi[tm].q[1] = *reinterpret_cast<const uint4*>(pa + 16);
            const unsigned short* pl = &AloS[cur][(waveM * 64 + tm * 16 + mrow) * LDAS + kbA];
            alo[tm].q[0] = *reinterpret_cast<const uint4*>(pl);
            alo[tm].q[1] = *reinterpret_cast<const uint4*>(pl + 16);
        }
        // B 32x16 bf16 layout: lanes 0-15 -> N=lane, K 0..7 & 16..23; lanes 16-31 -> K 8..15 & 24..31.
        const int kbB = (lane < 16) ? 0 : 8;
        #pragma unroll
        for (int tn = 0; tn < 2; ++tn) {
            const unsigned short* pb = &BhiS[cur][(waveN * 32 + tn * 16 + mrow) * LDAS + kbB];
            bhi[tn].q[0] = *reinterpret_cast<const uint4*>(pb);
            bhi[tn].q[1] = *reinterpret_cast<const uint4*>(pb + 16);
            const unsigned short* pl = &BloS[cur][(waveN * 32 + tn * 16 + mrow) * LDAS + kbB];
            blo[tn].q[0] = *reinterpret_cast<const uint4*>(pl);
            blo[tn].q[1] = *reinterpret_cast<const uint4*>(pl + 16);
        }

        // ---- 3-product split-precision accumulation: hi*hi + hi*lo + lo*hi ----
        #pragma unroll
        for (int tm = 0; tm < 4; ++tm) {
            #pragma unroll
            for (int tn = 0; tn < 2; ++tn) {
                acc[tm][tn] = __builtin_amdgcn_wmma_f32_16x16x32_bf16(
                    false, ahi[tm].v, false, bhi[tn].v, (short)0, acc[tm][tn], false, false);
                acc[tm][tn] = __builtin_amdgcn_wmma_f32_16x16x32_bf16(
                    false, ahi[tm].v, false, blo[tn].v, (short)0, acc[tm][tn], false, false);
                acc[tm][tn] = __builtin_amdgcn_wmma_f32_16x16x32_bf16(
                    false, alo[tm].v, false, bhi[tn].v, (short)0, acc[tm][tn], false, false);
            }
        }

        cur ^= 1;
    }

    // ---- epilogue: d2 = |x|^2 + |k|^2 - 2*dot ; out = exp(-max(d2,0)) + bias ----
    // C layout: VGPR r, lanes 0-15 -> M=r, lanes 16-31 -> M=r+8, N = lane%16.
    const int nloc   = lane & 15;
    const int rowAdd = (lane >> 4) * 8;
    #pragma unroll
    for (int tm = 0; tm < 4; ++tm) {
        #pragma unroll
        for (int tn = 0; tn < 2; ++tn) {
            int col   = n0 + waveN * 32 + tn * 16 + nloc;
            float kq  = ksq[col];
            float bv  = bias[col];
            int rbase = m0 + waveM * 64 + tm * 16 + rowAdd;
            #pragma unroll
            for (int r = 0; r < 8; ++r) {
                int row = rbase + r;
                float dot = acc[tm][tn][r];
                float d2  = fmaxf(xsq[row] + kq - 2.0f * dot, 0.0f);
                out[(size_t)row * U_DIM + col] = __expf(-d2) + bv;   // GAMMA = 1.0
            }
        }
    }
}

// ---------------------------------------------------------------------------
extern "C" void kernel_launch(void* const* d_in, const int* in_sizes, int n_in,
                              void* d_out, int out_size, void* d_ws, size_t ws_size,
                              hipStream_t stream) {
    const float* X    = (const float*)d_in[0];   // (8192, 512)
    const float* Kmat = (const float*)d_in[1];   // (512, 4096)
    const float* bias = (const float*)d_in[2];   // (4096,)
    float* out = (float*)d_out;                  // (8192, 4096)

    // Workspace layout (~24.05 MB):
    float* xsq = (float*)d_ws;                   // 8192 f
    float* ksq = xsq + B_DIM;                    // 4096 f
    unsigned short* Xhi  = (unsigned short*)(ksq + U_DIM);
    unsigned short* Xlo  = Xhi  + (size_t)B_DIM * D_DIM;
    unsigned short* KThi = Xlo  + (size_t)B_DIM * D_DIM;
    unsigned short* KTlo = KThi + (size_t)U_DIM * D_DIM;

    row_sq_norms<<<(B_DIM * 32) / 256, 256, 0, stream>>>(X, xsq);
    col_sq_norms<<<U_DIM / 256, 256, 0, stream>>>(Kmat, ksq);
    convert_x  <<<(B_DIM * D_DIM / 4) / 256, 256, 0, stream>>>(X, Xhi, Xlo);
    convert_k_t<<<(D_DIM * U_DIM / 4) / 256, 256, 0, stream>>>(Kmat, KThi, KTlo);

    dim3 grid(U_DIM / BN, B_DIM / BM);           // (32, 64)
    rbf_gemm<<<grid, 256, 0, stream>>>(Xhi, Xlo, KThi, KTlo, bias, xsq, ksq, out);
}
